// model_35064113004949
// MI455X (gfx1250) — compile-verified, matrix operand order checked by reading, not run on hardware
//
#include <hip/hip_runtime.h>
#include <hip/hip_bf16.h>

typedef _Float16 h16;
typedef __attribute__((ext_vector_type(16))) _Float16 v16h;
typedef __attribute__((ext_vector_type(8)))  _Float16 v8h;
typedef __attribute__((ext_vector_type(8)))  float    v8f;

constexpr int kB = 64, kT = 128, kN = 12, kNObj = 10;
constexpr int kFin = 12, kFenc = 64, kCoh = 14, kFx = 78;   // 64+14
constexpr int kDe = 128, kH = 128, kA = 16;
constexpr int kNBlk = 2 * kT * kB;                           // 16384

__device__ inline v8f wmma_f16(v16h a, v16h b, v8f c) {
  return __builtin_amdgcn_wmma_f32_16x16x32_f16(false, a, false, b, (short)0, c,
                                                false, false);
}

// k-offset (within a 32-wide kstep) for (lane, half-index) per CDNA5 16-bit
// A/B fragment layout: lanes 0-15 hold K {0..7,16..23}, lanes 16-31 are +8.
__device__ inline int frag_kk(int lane, int hidx) {
  int hi = lane >> 4;
  int v = hidx >> 1, pb = hidx & 1;
  int kkp = (v < 4) ? (2 * v + pb) : (16 + 2 * (v - 4) + pb);
  return kkp + hi * 8;
}

// ---------------- K0: swizzle a row-major f32 weight [K x N] into f16 WMMA
// B-operand fragments: frag[((j*nK + ks)*32 + lane)*16 + hidx] ----------------
__global__ void make_bfrag(const float* __restrict__ w, h16* __restrict__ frag,
                           int Kreal, int Ncols, int nK, int nJ) {
  int idx = blockIdx.x * 256 + threadIdx.x;
  int total = nJ * nK * 512;
  if (idx >= total) return;
  int hidx = idx & 15;
  int lane = (idx >> 4) & 31;
  int ks = (idx >> 9) % nK;
  int j = idx / (512 * nK);
  int col = j * 16 + (lane & 15);
  int k = ks * 32 + frag_kk(lane, hidx);
  float val = (k < Kreal && col < Ncols) ? w[k * Ncols + col] : 0.f;
  frag[idx] = (h16)val;
}

// ---------------- K1: encoder -> x[dir][t][b][n][80] f16 (78 real feats) ----
__global__ void encode_kernel(const float* __restrict__ obj,
                              const float* __restrict__ wrist,
                              const float* __restrict__ robj,
                              const float* __restrict__ rwrist,
                              const float* __restrict__ obj_ohs,
                              const float* __restrict__ wrist_ohs,
                              const float* __restrict__ w_obj,
                              const float* __restrict__ b_obj,
                              const float* __restrict__ w_hand,
                              const float* __restrict__ b_hand,
                              h16* __restrict__ x) {
  int idx = blockIdx.x * 256 + threadIdx.x;
  if (idx >= 2 * kT * kB * kN * 80) return;
  int f = idx % 80;
  int n = (idx / 80) % kN;
  int bb = (idx / (80 * kN)) % kB;
  int t = (idx / (80 * kN * kB)) % kT;
  int dir = idx / (80 * kN * kB * kT);
  float val = 0.f;
  if (f < kFenc) {
    const float *in, *w, *bias;
    if (n < kNObj) {
      in = (dir ? robj : obj) + ((bb * kT + t) * kNObj + n) * kFin;
      w = w_obj; bias = b_obj;
    } else {
      in = (dir ? rwrist : wrist) + ((bb * kT + t) * 2 + (n - kNObj)) * kFin;
      w = w_hand; bias = b_hand;
    }
    float acc = bias[f];
#pragma unroll
    for (int k = 0; k < kFin; ++k) acc += in[k] * w[k * kFenc + f];
    val = acc > 0.f ? acc : 0.f;
  } else if (f < kFx) {
    int c = f - kFenc;
    val = (n < kNObj) ? obj_ohs[(bb * kNObj + n) * kCoh + c]
                      : wrist_ohs[(bb * 2 + (n - kNObj)) * kCoh + c];
  }
  x[idx] = (h16)val;
}

// ---------------- K2: edge conv per (dir,t,b): pair GEMM (144x160x128) +
// relu + masked max over j, concat one-hots -> d fragments (1 Mtile x 5 ks) --
__global__ void __launch_bounds__(256) edge_kernel(
    const h16* __restrict__ x, const h16* __restrict__ wedgeF,
    const float* __restrict__ b_edge, const float* __restrict__ obj_ohs,
    const float* __restrict__ wrist_ohs, h16* __restrict__ dfrag) {
  __shared__ __align__(32) h16 xs[kN * 80];
  __shared__ __align__(32) h16 afrag[9 * 5 * 512];
  __shared__ float cbuf[144 * kDe];
  __shared__ float dred[kN * kDe];
  int blk = blockIdx.x;                 // (dir*T + t)*B + b
  int bb = blk % kB;
  int tid = threadIdx.x;
  int lane = tid & 31;
  int w = tid >> 5;

  for (int idx = tid; idx < kN * 80; idx += 256)
    xs[idx] = x[(size_t)blk * (kN * 80) + idx];
  __syncthreads();

  // Build pair-feature A matrix directly in fragment layout.
  for (int idx = tid; idx < 9 * 5 * 512; idx += 256) {
    int hidx = idx & 15;
    int ln = (idx >> 4) & 31;
    int ks = (idx >> 9) % 5;
    int m = idx / 2560;
    int row = m * 16 + (ln & 15);        // 0..143 (= i*12 + j)
    int k = ks * 32 + frag_kk(ln, hidx);
    int i = row / 12, j = row % 12;
    h16 val = (h16)0.f;
    if (k < kFx)            val = xs[i * 80 + k];
    else if (k < 2 * kFx)   val = (h16)(xs[j * 80 + (k - kFx)] -
                                        xs[i * 80 + (k - kFx)]);
    afrag[idx] = val;
  }
  __syncthreads();

  // 9x8 = 72 output tiles over 8 waves.
  for (int q = 0; q < 9; ++q) {
    int tt = q * 8 + w;
    int m = tt >> 3, j = tt & 7;
    v8f acc = {};
#pragma unroll
    for (int ks = 0; ks < 5; ++ks) {
      v16h a = *(const v16h*)&afrag[(m * 5 + ks) * 512 + lane * 16];
      v16h bm = *(const v16h*)&wedgeF[(j * 5 + ks) * 512 + lane * 16];
      acc = wmma_f16(a, bm, acc);
    }
    int col = j * 16 + (lane & 15);
    int mbase = m * 16 + ((lane >> 4) << 3);
#pragma unroll
    for (int r = 0; r < 8; ++r) cbuf[(mbase + r) * kDe + col] = acc[r];
  }
  __syncthreads();

  // relu(+bias), exclude diagonal, max over j (candidates >= 0 after relu).
  for (int idx = tid; idx < kN * kDe; idx += 256) {
    int i = idx / kDe, c = idx % kDe;
    float bias = b_edge[c];
    float mx = 0.f;
    for (int j = 0; j < kN; ++j) {
      if (j == i) continue;
      float v = cbuf[(i * kN + j) * kDe + c] + bias;
      if (v < 0.f) v = 0.f;
      if (v > mx) mx = v;
    }
    dred[idx] = mx;
  }
  __syncthreads();

  // Emit d = [maxpool(128), oh(14), pad] as A fragments (rows 12..15 zero).
  for (int idx = tid; idx < 5 * 512; idx += 256) {
    int hidx = idx & 15;
    int ln = (idx >> 4) & 31;
    int ks = idx >> 9;
    int row = ln & 15;
    int k = ks * 32 + frag_kk(ln, hidx);
    float val = 0.f;
    if (row < kN) {
      if (k < kDe) val = dred[row * kDe + k];
      else if (k < kDe + kCoh) {
        int c = k - kDe;
        val = (row < kNObj) ? obj_ohs[(bb * kNObj + row) * kCoh + c]
                            : wrist_ohs[(bb * 2 + (row - kNObj)) * kCoh + c];
      }
    }
    dfrag[(size_t)blk * 2560 + idx] = (h16)val;
  }
}

// ---------------- K3: gi = d @ w_ih + b_ih, all t in parallel (f16 frags) ---
__global__ void __launch_bounds__(256) gi_kernel(
    const h16* __restrict__ dfrag, const h16* __restrict__ wihF,
    const float* __restrict__ b_ih, h16* __restrict__ gi) {
  int blk = blockIdx.x;
  int tid = threadIdx.x;
  int lane = tid & 31;
  int w = tid >> 5;
  const h16* dp = dfrag + (size_t)blk * 2560;
  v16h a[5];
#pragma unroll
  for (int ks = 0; ks < 5; ++ks) a[ks] = *(const v16h*)&dp[ks * 512 + lane * 16];
  for (int q = 0; q < 3; ++q) {
    int j = q * 8 + w;                   // 0..23
    v8f acc = {};
#pragma unroll
    for (int ks = 0; ks < 5; ++ks) {
      v16h bm = *(const v16h*)&wihF[(j * 5 + ks) * 512 + lane * 16];
      acc = wmma_f16(a[ks], bm, acc);
    }
    float bias = b_ih[j * 16 + (lane & 15)];
    v8h out;
#pragma unroll
    for (int r = 0; r < 8; ++r) out[r] = (h16)(acc[r] + bias);
    *(v8h*)&gi[((size_t)blk * 24 + j) * 256 + lane * 8] = out;
  }
}

// ---------------- K4: h0 = relu(d0 @ w_hi + b_hi), stored as f32 C-frags ----
__global__ void __launch_bounds__(128) h0_kernel(
    const h16* __restrict__ dfrag, const h16* __restrict__ whiF,
    const float* __restrict__ b_hi, float* __restrict__ h0) {
  int blk = blockIdx.x;                 // dir*64 + b
  int dir = blk >> 6, bb = blk & 63;
  int tid = threadIdx.x;
  int lane = tid & 31;
  int w = tid >> 5;
  const h16* dp = dfrag + (size_t)(dir * kT * kB + bb) * 2560;   // t = 0
  v16h a[5];
#pragma unroll
  for (int ks = 0; ks < 5; ++ks) a[ks] = *(const v16h*)&dp[ks * 512 + lane * 16];
  for (int q = 0; q < 2; ++q) {
    int j = q * 4 + w;                   // 0..7
    v8f acc = {};
#pragma unroll
    for (int ks = 0; ks < 5; ++ks) {
      v16h bm = *(const v16h*)&whiF[(j * 5 + ks) * 512 + lane * 16];
      acc = wmma_f16(a[ks], bm, acc);
    }
    float bias = b_hi[j * 16 + (lane & 15)];
#pragma unroll
    for (int r = 0; r < 8; ++r) {
      float v = acc[r] + bias;
      acc[r] = v > 0.f ? v : 0.f;
    }
    *(v8f*)&h0[((size_t)blk * 8 + j) * 256 + lane * 8] = acc;
  }
}

// Restage a C-layout f32 tile of h into the per-wave A-fragment LDS buffer.
__device__ inline void stage_tile(h16* stw, int lane, int j, v8f hv) {
  int col = j * 16 + (lane & 15);        // k-dim for next step's GEMM
  int mbase = (lane >> 4) << 3;
  int ks = col >> 5;
  int kk = col & 31;
  int hi2 = (kk >> 3) & 1;
  int kkp = kk - hi2 * 8;
  int v2 = (kkp < 8) ? (kkp >> 1) : (4 + ((kkp - 16) >> 1));
  int hidx = v2 * 2 + (kkp & 1);
#pragma unroll
  for (int r = 0; r < 8; ++r) {
    int lane_t = (mbase + r) + hi2 * 16;
    stw[(ks * 32 + lane_t) * 16 + hidx] = (h16)hv[r];
  }
}

// ---------------- K5: persistent GRU. One wave owns one (dir,b) -> 12 live
// rows, loops t=0..127 with two block barriers per step; w_hh fragments in LDS. -----
__global__ void __launch_bounds__(128) gru_kernel(
    const h16* __restrict__ gi, const h16* __restrict__ whhF,
    const float* __restrict__ b_hh, const float* __restrict__ h0,
    float* __restrict__ hw) {
  __shared__ __align__(32) h16 whhs[24 * 4 * 512];   // 96 KB
  __shared__ __align__(32) h16 stag[4 * 2048];       // 4 KB per wave
  int tid = threadIdx.x;
  int lane = tid & 31;
  int w = tid >> 5;
  int gw = blockIdx.x * 4 + w;           // 0..127 = dir*64 + b
  int dir = gw >> 6, bb = gw & 63;

  for (int idx = tid; idx < 24 * 4 * 512 / 8; idx += 128)
    ((uint4*)whhs)[idx] = ((const uint4*)whhF)[idx];

  h16* stw = &stag[w * 2048];
  v8f h[8];
#pragma unroll
  for (int j = 0; j < 8; ++j)
    h[j] = *(const v8f*)&h0[((size_t)gw * 8 + j) * 256 + lane * 8];
  __syncthreads();
#pragma unroll
  for (int j = 0; j < 8; ++j) stage_tile(stw, lane, j, h[j]);
  __syncthreads();
  v16h afr[4];
#pragma unroll
  for (int k = 0; k < 4; ++k) afr[k] = *(const v16h*)&stw[k * 512 + lane * 16];
  __syncthreads();

  int colL = lane & 15;
  for (int t = 0; t < kT; ++t) {
    const h16* gip = gi + (size_t)((dir * kT + t) * kB + bb) * (24 * 256);
    for (int j = 0; j < 8; ++j) {
      v8f ar = {}, az = {}, an = {};
#pragma unroll
      for (int k = 0; k < 4; ++k) {
        v16h br = *(const v16h*)&whhs[(j * 4 + k) * 512 + lane * 16];
        v16h bz = *(const v16h*)&whhs[((j + 8) * 4 + k) * 512 + lane * 16];
        v16h bn = *(const v16h*)&whhs[((j + 16) * 4 + k) * 512 + lane * 16];
        ar = wmma_f16(afr[k], br, ar);
        az = wmma_f16(afr[k], bz, az);
        an = wmma_f16(afr[k], bn, an);
      }
      int col = j * 16 + colL;
      float bhr = b_hh[col], bhz = b_hh[kH + col], bhn = b_hh[2 * kH + col];
      v8h gir = *(const v8h*)&gip[(j * 32 + lane) * 8];
      v8h giz = *(const v8h*)&gip[((j + 8) * 32 + lane) * 8];
      v8h gin = *(const v8h*)&gip[((j + 16) * 32 + lane) * 8];
#pragma unroll
      for (int r = 0; r < 8; ++r) {
        float rr = 1.f / (1.f + __expf(-((float)gir[r] + ar[r] + bhr)));
        float zz = 1.f / (1.f + __expf(-((float)giz[r] + az[r] + bhz)));
        float nn = tanhf((float)gin[r] + rr * (an[r] + bhn));
        h[j][r] = (1.f - zz) * nn + zz * h[j][r];
      }
      // Rows 10,11 (wrists) live in VGPR slots 2,3 of lanes 16-31.
      if (lane >= 16) {
        float* hwp = hw + ((size_t)((dir * kT + t) * kB + bb) * 2) * kH;
        hwp[col] = h[j][2];
        hwp[kH + col] = h[j][3];
      }
      stage_tile(stw, lane, j, h[j]);
    }
    __syncthreads();
#pragma unroll
    for (int k = 0; k < 4; ++k)
      afr[k] = *(const v16h*)&stw[k * 512 + lane * 16];
    __syncthreads();
  }
}

// ---------------- K6: interleave+tanh of (hf,hr) wrist states, classifier ---
__global__ void __launch_bounds__(64) clf_kernel(
    const float* __restrict__ hw, const float* __restrict__ w_lclf,
    const float* __restrict__ b_lclf, const float* __restrict__ w_rclf,
    const float* __restrict__ b_rclf, float* __restrict__ out) {
  __shared__ float th[4 * kH];           // [(node*2 + dir)][k]
  int blk = blockIdx.x;                  // t*B + b
  int t = blk / kB, bb = blk % kB;
  int tid = threadIdx.x;
  for (int idx = tid; idx < 4 * kH; idx += 64) {
    int node = idx >> 8;
    int dir = (idx >> 7) & 1;
    int k = idx & 127;
    th[idx] = tanhf(hw[((size_t)((dir * kT + t) * kB + bb) * 2 + node) * kH + k]);
  }
  __syncthreads();
  if (tid < 32) {
    int hand = tid >> 4;                 // 0 = left (node 10), 1 = right (11)
    int a = tid & 15;
    const float* wclf = hand ? w_rclf : w_lclf;
    float acc = hand ? b_rclf[a] : b_lclf[a];
    const float* th0 = &th[(hand * 2 + 0) * kH];   // fwd
    const float* th1 = &th[(hand * 2 + 1) * kH];   // rev
    for (int k = 0; k < kH; ++k)
      acc += th0[k] * wclf[(2 * k) * kA + a] + th1[k] * wclf[(2 * k + 1) * kA + a];
    out[((size_t)(hand * kB + bb) * kT + t) * kA + a] = acc;
  }
}

extern "C" void kernel_launch(void* const* d_in, const int* in_sizes, int n_in,
                              void* d_out, int out_size, void* d_ws,
                              size_t ws_size, hipStream_t stream) {
  (void)in_sizes; (void)n_in; (void)out_size; (void)ws_size;
  const float* obj       = (const float*)d_in[0];
  const float* wrist     = (const float*)d_in[1];
  const float* robj      = (const float*)d_in[2];
  const float* rwrist    = (const float*)d_in[3];
  const float* obj_ohs   = (const float*)d_in[4];
  const float* wrist_ohs = (const float*)d_in[5];
  const float* w_obj     = (const float*)d_in[6];
  const float* b_obj     = (const float*)d_in[7];
  const float* w_hand    = (const float*)d_in[8];
  const float* b_hand    = (const float*)d_in[9];
  const float* w_edge    = (const float*)d_in[10];
  const float* b_edge    = (const float*)d_in[11];
  const float* w_hi      = (const float*)d_in[12];
  const float* b_hi      = (const float*)d_in[13];
  const float* w_ih      = (const float*)d_in[14];
  const float* w_hh      = (const float*)d_in[15];
  const float* b_ih      = (const float*)d_in[16];
  const float* b_hh      = (const float*)d_in[17];
  const float* w_lclf    = (const float*)d_in[18];
  const float* b_lclf    = (const float*)d_in[19];
  const float* w_rclf    = (const float*)d_in[20];
  const float* b_rclf    = (const float*)d_in[21];
  float* out = (float*)d_out;

  char* p = (char*)d_ws;
  h16* x      = (h16*)p; p += (size_t)2 * kT * kB * kN * 80 * 2;   // 31.5 MB
  h16* wedgeF = (h16*)p; p += 8 * 5 * 512 * 2;
  h16* wihF   = (h16*)p; p += 24 * 5 * 512 * 2;
  h16* whiF   = (h16*)p; p += 8 * 5 * 512 * 2;
  h16* whhF   = (h16*)p; p += 24 * 4 * 512 * 2;
  h16* dfrag  = (h16*)p; p += (size_t)kNBlk * 2560 * 2;            // 84 MB
  h16* gi     = (h16*)p; p += (size_t)kNBlk * 24 * 256 * 2;        // 201 MB
  float* h0   = (float*)p; p += (size_t)128 * 8 * 256 * 4;         // 1 MB
  float* hw   = (float*)p; p += (size_t)2 * kT * kB * 2 * kH * 4;  // 17 MB

  make_bfrag<<<(8 * 5 * 512 + 255) / 256, 256, 0, stream>>>(w_edge, wedgeF, 156, 128, 5, 8);
  make_bfrag<<<(24 * 5 * 512 + 255) / 256, 256, 0, stream>>>(w_ih, wihF, 142, 384, 5, 24);
  make_bfrag<<<(8 * 5 * 512 + 255) / 256, 256, 0, stream>>>(w_hi, whiF, 142, 128, 5, 8);
  make_bfrag<<<(24 * 4 * 512 + 255) / 256, 256, 0, stream>>>(w_hh, whhF, 128, 384, 4, 24);
  encode_kernel<<<(2 * kT * kB * kN * 80 + 255) / 256, 256, 0, stream>>>(
      obj, wrist, robj, rwrist, obj_ohs, wrist_ohs, w_obj, b_obj, w_hand, b_hand, x);
  edge_kernel<<<kNBlk, 256, 0, stream>>>(x, wedgeF, b_edge, obj_ohs, wrist_ohs, dfrag);
  gi_kernel<<<kNBlk, 256, 0, stream>>>(dfrag, wihF, b_ih, gi);
  h0_kernel<<<128, 128, 0, stream>>>(dfrag, whiF, b_hi, h0);
  gru_kernel<<<32, 128, 0, stream>>>(gi, whhF, b_hh, h0, hw);
  clf_kernel<<<kT * kB, 64, 0, stream>>>(hw, w_lclf, b_lclf, w_rclf, b_rclf, out);
}
